// SparseSelfAttention_45114336477584
// MI455X (gfx1250) — compile-verified, matrix operand order checked by reading, not run on hardware
//
#include <hip/hip_runtime.h>
#include <hip/hip_bf16.h>

// ---- problem constants (match reference) ----
#define S_LEN   2048
#define DM      1024
#define NH      16
#define HD      64
#define BATCH   2
#define MROWS   (BATCH * S_LEN)                       // 4096 rows of x
#define XH_ELEMS  ((size_t)MROWS * DM)                // 4,194,304
#define WT_ELEMS  ((size_t)DM * DM)                   // 1,048,576 per projection
#define QKV_ELEMS ((size_t)BATCH * NH * S_LEN * HD)   // 4,194,304 per tensor

#define LDS_PITCH 40                                  // halves; 80B row pitch (bank-conflict-free)

typedef _Float16 v16h __attribute__((ext_vector_type(16)));
typedef _Float16 v8h  __attribute__((ext_vector_type(8)));
typedef float    v8f  __attribute__((ext_vector_type(8)));
typedef float    f32x4 __attribute__((ext_vector_type(4)));

// Fragment K-offset map for 16-bit WMMA operands (ISA 7.12.2, 16-bit A 16x32):
//   lane<16 : element i -> K = i + (i>=8 ? 8 : 0)   ; lane>=16: same + 8
// => every fragment is two contiguous 8-half (16B) runs at K = half*8 and 16+half*8.

__device__ __forceinline__ v16h load_frag16(const _Float16* p /* base + half*8 */) {
    v8h lo = *(const v8h*)p;
    v8h hi = *(const v8h*)(p + 16);
    v16h f;
#pragma unroll
    for (int i = 0; i < 8; ++i) { f[i] = lo[i]; f[8 + i] = hi[i]; }
    return f;
}

// ---- CDNA5 async global->LDS DMA (CDNA5 ISA; tracked by ASYNCcnt) ----
__device__ __forceinline__ void async_load_b128(void* lds_ptr, const void* gptr) {
    const unsigned loff = (unsigned)(size_t)lds_ptr;   // low 32 bits = LDS offset
    const unsigned long long ga = (unsigned long long)(size_t)gptr;
    asm volatile("global_load_async_to_lds_b128 %0, %1, off"
                 :: "v"(loff), "v"(ga) : "memory");
}
__device__ __forceinline__ void wait_asynccnt0() {
    asm volatile("s_wait_asynccnt 0x0" ::: "memory");
}

// ============================================================================
// Kernel 0a: x (f32) -> xh (f16), elementwise, 8 elems/thread, b128 loads.
// ============================================================================
__global__ __launch_bounds__(256)
void cvt_x_f16(const float* __restrict__ x, _Float16* __restrict__ xh)
{
    const size_t i0 = ((size_t)blockIdx.x * 256 + threadIdx.x) * 8;
    f32x4 a = *(const f32x4*)(x + i0);
    f32x4 b = *(const f32x4*)(x + i0 + 4);
    v8h o;
#pragma unroll
    for (int i = 0; i < 4; ++i) { o[i] = (_Float16)a[i]; o[4 + i] = (_Float16)b[i]; }
    *(v8h*)(xh + i0) = o;
}

// ============================================================================
// Kernel 0b: W (f32, [k][n]) -> Wt (f16, [n][k]) via LDS 64x64 tile transpose.
// ============================================================================
__global__ __launch_bounds__(256)
void cvt_w_transpose(const float* __restrict__ Wq, const float* __restrict__ Wk,
                     const float* __restrict__ Wv, _Float16* __restrict__ wt)
{
    __shared__ _Float16 tile[64][65];
    const float* W = (blockIdx.z == 0) ? Wq : (blockIdx.z == 1) ? Wk : Wv;
    _Float16* out = wt + (size_t)blockIdx.z * WT_ELEMS;

    const int n0 = blockIdx.x * 64, k0 = blockIdx.y * 64;
    const int c = threadIdx.x & 63, rb = threadIdx.x >> 6;
#pragma unroll
    for (int jj = 0; jj < 16; ++jj) {
        const int r = rb + jj * 4;
        tile[r][c] = (_Float16)W[(size_t)(k0 + r) * DM + n0 + c];
    }
    __syncthreads();
#pragma unroll
    for (int jj = 0; jj < 16; ++jj) {
        const int rn = rb + jj * 4;
        out[(size_t)(n0 + rn) * DM + k0 + c] = tile[c][rn];
    }
}

// ============================================================================
// Kernel 1: QKV GEMM. 256-thread block = 8 waves own a 128x128 tile.
// Per 32-K step: async-DMA 8KB A + 8KB B into LDS (double-buffered),
// each wave computes a 32x64 subtile = 8 WMMAs from LDS fragments.
// Q,K stored f16 [b,h,s,d]; V stored f16 TRANSPOSED [b,h,d,s].
// ============================================================================
__device__ __forceinline__ void stage_tiles(const _Float16* __restrict__ xh,
                                            const _Float16* __restrict__ w,
                                            int blockM, int blockN, int k0,
                                            _Float16 (*As)[LDS_PITCH],
                                            _Float16 (*Bs)[LDS_PITCH], int tid)
{
    const int row = tid >> 1;            // 0..127
    const int c0  = (tid & 1) * 16;      // half-offset within the 32-wide K slab
    const _Float16* ga = xh + (size_t)(blockM + row) * DM + k0 + c0;
    const _Float16* gb = w  + (size_t)(blockN + row) * DM + k0 + c0;
    async_load_b128(&As[row][c0],     ga);
    async_load_b128(&As[row][c0 + 8], ga + 8);
    async_load_b128(&Bs[row][c0],     gb);
    async_load_b128(&Bs[row][c0 + 8], gb + 8);
}

__global__ __launch_bounds__(256)
void qkv_gemm_wmma(const _Float16* __restrict__ xh, const _Float16* __restrict__ wt,
                   const float* __restrict__ bq, const float* __restrict__ bk,
                   const float* __restrict__ bv, _Float16* __restrict__ qkv)
{
    __shared__ _Float16 As[2][128][LDS_PITCH];
    __shared__ _Float16 Bs[2][128][LDS_PITCH];

    const int tid  = threadIdx.x;
    const int lane = tid & 31, wv = tid >> 5;       // 8 waves
    const int wm = wv >> 1, wn = wv & 1;            // 4(M) x 2(N) wave grid
    const int half = lane >> 4, r16 = lane & 15;
    const int blockM = (blockIdx.x >> 3) * 128;     // 32 m-blocks
    const int blockN = (blockIdx.x & 7) * 128;      // 8 n-blocks
    const int proj = blockIdx.y;

    const _Float16* wmat = wt + (size_t)proj * WT_ELEMS;
    const float* bias = (proj == 0) ? bq : (proj == 1) ? bk : bv;

    v8f acc[2][4] = {{{}, {}, {}, {}}, {{}, {}, {}, {}}};

    stage_tiles(xh, wmat, blockM, blockN, 0, As[0], Bs[0], tid);

    int buf = 0;
    for (int ks = 0; ks < DM / 32; ++ks) {
        wait_asynccnt0();            // my DMA into As/Bs[buf] landed
        __syncthreads();             // everyone's landed; prior reads of buf^1 done
        if (ks + 1 < DM / 32)
            stage_tiles(xh, wmat, blockM, blockN, (ks + 1) * 32,
                        As[buf ^ 1], Bs[buf ^ 1], tid);

        v16h a[2], bf[4];
#pragma unroll
        for (int i = 0; i < 2; ++i)
            a[i] = load_frag16(&As[buf][wm * 32 + 16 * i + r16][half * 8]);
#pragma unroll
        for (int j = 0; j < 4; ++j)
            bf[j] = load_frag16(&Bs[buf][wn * 64 + 16 * j + r16][half * 8]);
#pragma unroll
        for (int i = 0; i < 2; ++i)
#pragma unroll
            for (int j = 0; j < 4; ++j)
                acc[i][j] = __builtin_amdgcn_wmma_f32_16x16x32_f16(
                    false, a[i], false, bf[j], (short)0, acc[i][j], false, false);
        buf ^= 1;
    }

    _Float16* outp = qkv + (size_t)proj * QKV_ELEMS;
    const int m0 = blockM + wm * 32;
#pragma unroll
    for (int j = 0; j < 4; ++j) {
        const int n = blockN + wn * 64 + 16 * j + r16;
        const float bb = bias[n];
        const int h = n >> 6, d = n & 63;
        if (proj == 2) {
            // V transposed: [b,h,d,s]; s contiguous -> one 16B store per subtile
#pragma unroll
            for (int i = 0; i < 2; ++i) {
                const int m = m0 + 16 * i + half * 8;   // aligned run of 8, no batch cross
                const int b = m >> 11, s = m & (S_LEN - 1);
                v8h vvv;
#pragma unroll
                for (int rr = 0; rr < 8; ++rr) vvv[rr] = (_Float16)(acc[i][j][rr] + bb);
                *(v8h*)(outp + ((size_t)(b * NH + h) * HD + d) * S_LEN + s) = vvv;
            }
        } else {
            // Q,K: [b,h,s,d]; scatter b16 stores (epilogue only)
#pragma unroll
            for (int i = 0; i < 2; ++i)
#pragma unroll
                for (int rr = 0; rr < 8; ++rr) {
                    const int m = m0 + 16 * i + half * 8 + rr;
                    const int b = m >> 11, s = m & (S_LEN - 1);
                    outp[((size_t)(b * NH + h) * S_LEN + s) * HD + d] =
                        (_Float16)(acc[i][j][rr] + bb);
                }
        }
    }
}

// ============================================================================
// Kernel 2: block-sparse flash attention, 2 q-block columns per wave.
// The fixed layout means both columns share every global k-block and overlap
// on local blocks -> K/V/mask loads amortized 2x (~190 MB total traffic).
// S^T = K_blk * Q^T : softmax k-reduction = 8 regs + shfl_xor(16).
// O^T += V^T * P^T  : P^T C-frag aliases the B-frag register-for-register.
// Column participation (qb >= kb) is wave-uniform -> EXEC stays all-1s.
// ============================================================================
__global__ __launch_bounds__(128)
void sparse_attn_wmma(const _Float16* __restrict__ qkv,
                      const float* __restrict__ kpm,
                      float* __restrict__ out)
{
    const int lane = threadIdx.x & 31;
    const int wid  = (blockIdx.x << 2) | (threadIdx.x >> 5); // 0..2047
    const int pr  = wid & 1;            // which column pair of the window
    const int win = (wid >> 1) & 31;    // 32 windows of 4 q-blocks
    const int bh  = wid >> 6;           // 0..31
    const int b = bh >> 4, h = bh & 15;
    const int half = lane >> 4, r16 = lane & 15;

    const _Float16* qbuf = qkv;                       // [b,h,s,d]
    const _Float16* kbuf = qkv + QKV_ELEMS;           // [b,h,s,d]
    const _Float16* vt   = qkv + 2 * QKV_ELEMS;       // [b,h,d,s]
    const size_t head_qk = (size_t)bh * S_LEN * HD;
    const size_t head_v  = (size_t)bh * HD * S_LEN;

    const int w0  = win * 4;            // window start block
    const int qb0 = w0 + 2 * pr;        // this wave's columns: qb0, qb0+1

    // Q^T B-fragments for both columns (fixed per wave)
    v16h bq[2][2];
#pragma unroll
    for (int c = 0; c < 2; ++c) {
        const _Float16* pq = qbuf + head_qk + (size_t)((qb0 + c) * 16 + r16) * HD + half * 8;
        bq[c][0] = load_frag16(pq);
        bq[c][1] = load_frag16(pq + 32);
    }

    v8f o[2][4] = {{{}, {}, {}, {}}, {{}, {}, {}, {}}};
    float mrun[2] = {-__builtin_inff(), -__builtin_inff()};
    float lrun[2] = {0.f, 0.f};

    const int ng = win;                 // global blocks k = 3,7,... < w0
    const int nl = 2 * pr + 2;          // local blocks w0 .. qb0+1

    for (int t = 0; t < ng + nl; ++t) {
        const int kb = (t < ng) ? (t * 4 + 3) : (w0 + (t - ng));
        const int jl = t - ng;          // local block index (>=0 once local)

        // K-block A-fragments (shared by both columns)
        const _Float16* pk = kbuf + head_qk + (size_t)(kb * 16 + r16) * HD + half * 8;
        const v16h ak0 = load_frag16(pk);
        const v16h ak1 = load_frag16(pk + 32);

        // key-padding mask (per k token = per C row)
        const float* pm = kpm + b * S_LEN + kb * 16 + half * 8;
        const f32x4 pm0 = *(const f32x4*)pm;
        const f32x4 pm1 = *(const f32x4*)(pm + 4);

        // V^T A-fragments for this k-block (shared by both columns; K pad 16->32)
        const _Float16* pv = vt + head_v + (size_t)r16 * S_LEN + kb * 16 + half * 8;
        v16h av[4];
#pragma unroll
        for (int dc = 0; dc < 4; ++dc) {
            const v8h vlo = *(const v8h*)(pv + (size_t)dc * 16 * S_LEN);
#pragma unroll
            for (int i = 0; i < 8; ++i) { av[dc][i] = vlo[i]; av[dc][8 + i] = (_Float16)0.f; }
        }

#pragma unroll
        for (int c = 0; c < 2; ++c) {
            if (t < ng || 2 * pr + c >= jl) {    // wave-uniform block-causal guard
                v8f st = {};
                st = __builtin_amdgcn_wmma_f32_16x16x32_f16(false, ak0, false, bq[c][0], (short)0, st, false, false);
                st = __builtin_amdgcn_wmma_f32_16x16x32_f16(false, ak1, false, bq[c][1], (short)0, st, false, false);
#pragma unroll
                for (int rr = 0; rr < 4; ++rr) {
                    st[rr]     = st[rr]     * 0.125f + pm0[rr];
                    st[4 + rr] = st[4 + rr] * 0.125f + pm1[rr];
                }
                // online softmax: column(q) max over 8 regs + swap half-waves
                float bm = st[0];
#pragma unroll
                for (int rr = 1; rr < 8; ++rr) bm = fmaxf(bm, st[rr]);
                bm = fmaxf(bm, __shfl_xor(bm, 16, 32));
                const float mnew  = fmaxf(mrun[c], bm);
                const float alpha = __expf(mrun[c] - mnew);

                v16h bp;                         // P^T C-frag -> B-frag, register-aliased
                float bs = 0.f;
#pragma unroll
                for (int rr = 0; rr < 8; ++rr) {
                    const float p = __expf(st[rr] - mnew);
                    bs += p;
                    bp[rr]     = (_Float16)p;
                    bp[8 + rr] = (_Float16)0.f;
                }
                bs += __shfl_xor(bs, 16, 32);
                lrun[c] = lrun[c] * alpha + bs;
                mrun[c] = mnew;

#pragma unroll
                for (int dc = 0; dc < 4; ++dc)
#pragma unroll
                    for (int rr = 0; rr < 8; ++rr) o[c][dc][rr] *= alpha;

#pragma unroll
                for (int dc = 0; dc < 4; ++dc)
                    o[c][dc] = __builtin_amdgcn_wmma_f32_16x16x32_f16(
                        false, av[dc], false, bp, (short)0, o[c][dc], false, false);
            }
        }
    }

    // normalize and store both columns: row d = dc*16 + rr + half*8, col q = r16
#pragma unroll
    for (int c = 0; c < 2; ++c) {
        const float rl = 1.0f / lrun[c];
        float* orow = out + ((size_t)b * S_LEN + (qb0 + c) * 16 + r16) * DM + h * HD;
#pragma unroll
        for (int dc = 0; dc < 4; ++dc)
#pragma unroll
            for (int rr = 0; rr < 8; ++rr)
                orow[dc * 16 + rr + half * 8] = o[c][dc][rr] * rl;
    }
}

// ============================================================================
extern "C" void kernel_launch(void* const* d_in, const int* in_sizes, int n_in,
                              void* d_out, int out_size, void* d_ws, size_t ws_size,
                              hipStream_t stream) {
    const float* x   = (const float*)d_in[0];
    const float* kpm = (const float*)d_in[1];
    const float* Wq  = (const float*)d_in[2];
    const float* bq  = (const float*)d_in[3];
    const float* Wk  = (const float*)d_in[4];
    const float* bk  = (const float*)d_in[5];
    const float* Wv  = (const float*)d_in[6];
    const float* bv  = (const float*)d_in[7];
    float* out = (float*)d_out;

    // workspace layout (f16): xh | Wt(x3) | Q | K | Vt   = 8+6+24 = 38 MB
    _Float16* xh  = (_Float16*)d_ws;
    _Float16* wt  = xh + XH_ELEMS;
    _Float16* qkv = wt + 3 * WT_ELEMS;

    cvt_x_f16<<<dim3(XH_ELEMS / (256 * 8)), 256, 0, stream>>>(x, xh);
    cvt_w_transpose<<<dim3(DM / 64, DM / 64, 3), 256, 0, stream>>>(Wq, Wk, Wv, wt);

    // per projection: 32 m-blocks x 8 n-blocks of 128x128; 8 waves/block
    qkv_gemm_wmma<<<dim3(32 * 8, 3), 256, 0, stream>>>(xh, wt, bq, bk, bv, qkv);

    // 2 batches * 16 heads * 32 windows * 2 column-pairs = 2048 waves, 4/block
    sparse_attn_wmma<<<dim3(2048 / 4), 128, 0, stream>>>(qkv, kpm, out);
}
